// ChunkedBilinear_69664369541479
// MI455X (gfx1250) — matrix operand order repaired and checked by reading, not compiled, not run on hardware
//
#include <hip/hip_runtime.h>

typedef _Float16 half_t;
typedef __attribute__((ext_vector_type(16))) _Float16 v16h;
typedef __attribute__((ext_vector_type(8)))  float    v8f;

// Problem constants (match reference setup_inputs: B=4, L=1024, F=768, O=32, E=20, C=128)
#define BB   4
#define LL   1024
#define FF   768
#define OO   32
#define EE   20
#define FIN  788
#define CC   128
#define NCH  8      // L/C
#define KK   15     // 2*NCH-1
#define NG   9      // nbins+1 (bins 2..256 -> 8 bins)
#define NPG  288    // OO*NG

// ---------------------------------------------------------------------------
// f32 -> f16 convert (two arrays at once)
// ---------------------------------------------------------------------------
__global__ void convert_f16_kernel(const float* __restrict__ a, half_t* __restrict__ oa,
                                   const float* __restrict__ b, half_t* __restrict__ ob,
                                   size_t n) {
  size_t i = (size_t)blockIdx.x * blockDim.x + threadIdx.x;
  size_t stride = (size_t)gridDim.x * blockDim.x;
  for (; i < n; i += stride) {
    oa[i] = (half_t)a[i];
    ob[i] = (half_t)b[i];
  }
}

// ---------------------------------------------------------------------------
// WffT[o][j][i] = (f16) weight[o][i][j]   (i,j < 768) — LDS tile transpose
// grid (24,24,32), block 256
// ---------------------------------------------------------------------------
__global__ __launch_bounds__(256)
void wffT_kernel(const float* __restrict__ w, half_t* __restrict__ out) {
  __shared__ float tile[32][33];
  const int o  = blockIdx.z;
  const int i0 = blockIdx.y * 32;
  const int j0 = blockIdx.x * 32;
  const int tx = threadIdx.x & 31;
  const int ty = threadIdx.x >> 5;  // 0..7
  const float* wo = w + (size_t)o * FIN * FIN;
#pragma unroll
  for (int p = 0; p < 4; ++p) {
    int i = i0 + ty + p * 8;
    tile[ty + p * 8][tx] = wo[(size_t)i * FIN + (j0 + tx)];
  }
  __syncthreads();
  half_t* oo = out + (size_t)o * FF * FF;
#pragma unroll
  for (int p = 0; p < 4; ++p) {
    int j = j0 + ty + p * 8;
    oo[(size_t)j * FF + (i0 + tx)] = (half_t)tile[tx][ty + p * 8];
  }
}

// ---------------------------------------------------------------------------
// v2t[(o*9+g)][i] = sum_e Wfe[o,i,e]*to_emb[g,e]
// v3t[(o*9+g)][j] = sum_i from_emb[g,i]*Wef[o,i,j]
// ---------------------------------------------------------------------------
__global__ void v2v3_kernel(const float* __restrict__ w, const float* __restrict__ fe,
                            const float* __restrict__ te, half_t* __restrict__ v2t,
                            half_t* __restrict__ v3t) {
  int idx = blockIdx.x * 256 + threadIdx.x;
  if (idx >= NPG * FF) return;
  int n = idx / FF, i = idx - n * FF;
  int o = n / NG, g = n - o * NG;
  const float* wo = w + (size_t)o * FIN * FIN;
  float s2 = 0.f, s3 = 0.f;
#pragma unroll
  for (int e = 0; e < EE; ++e) {
    s2 += wo[(size_t)i * FIN + FF + e] * te[g * EE + e];
    s3 += fe[g * EE + e] * wo[(size_t)(FF + e) * FIN + i];
  }
  v2t[idx] = (half_t)s2;
  v3t[idx] = (half_t)s3;
}

// ---------------------------------------------------------------------------
// T4[g*32+o] = from_emb[g] . Wee[o] . to_emb[g] + bias[o]
// ---------------------------------------------------------------------------
__global__ void t4_kernel(const float* __restrict__ w, const float* __restrict__ bias,
                          const float* __restrict__ fe, const float* __restrict__ te,
                          float* __restrict__ T4) {
  int t = threadIdx.x;
  if (t >= NPG) return;
  int g = t >> 5, o = t & 31;
  const float* wo = w + (size_t)o * FIN * FIN;
  float s = 0.f;
  for (int i = 0; i < EE; ++i) {
    float inner = 0.f;
#pragma unroll
    for (int e = 0; e < EE; ++e)
      inner += wo[(size_t)(FF + i) * FIN + FF + e] * te[g * EE + e];
    s += fe[g * EE + i] * inner;
  }
  T4[t] = s + bias[o];
}

// ---------------------------------------------------------------------------
// Templated f16 GEMM, D[m,n] = sum_k A[m,k] * Bt[n,k], f32 accumulate via
// v_wmma_f32_16x16x32_f16. Block tile 128x128, 8 waves (wave32), each wave
// 64x32 = 4x2 WMMA tiles. K fixed at 768 (24 steps of 32).
// Double-buffered LDS (80B padded rows, conflict-free ds_load_b128), filled
// with global_load_async_to_lds_b128 (ASYNCcnt) overlapping the WMMA work.
//
// MODE 0: stage1  z=(m<<5)|o : A=Fh_b+m*C*K, Bt=WffT+o*768*K, out(f16)=Gf+z*C*768
// MODE 1: stage2  z=(k<<5)|o : A=Gf+((k/2)*32+o)*C*K, Bt=Th_b+((k+1)/2)*C*K,
//                              out(f32)=Ytmp+z*C*C
// MODE 2: P2/P3   out(f32), ldc=NPG; B rows >= NPG are clamped (their garbage
//                 only reaches output columns >= NPG, which are never stored)
// ---------------------------------------------------------------------------
template <int MODE>
__global__ __launch_bounds__(256)
void gemm_tpl(const half_t* __restrict__ Abase, const half_t* __restrict__ Bbase,
              void* __restrict__ Obase) {
  constexpr int KD = FF;            // 768
  constexpr int NK = KD / 32;       // 24
  __shared__ __align__(16) half_t As[2][128][40];
  __shared__ __align__(16) half_t Bs[2][128][40];
  constexpr unsigned BUFB = 128u * 40u * (unsigned)sizeof(half_t);  // 10240

  const int tid   = threadIdx.x;
  const int lane  = tid & 31;
  const int wv    = tid >> 5;
  const int waveM = wv >> 2;   // 0..1
  const int waveN = wv & 3;    // 0..3
  const int mrow  = lane & 15;
  const int sel   = lane >> 4; // 0..1

  const int nBase = blockIdx.x * 128;
  const int mBase = blockIdx.y * 128;
  const int z     = blockIdx.z;

  const half_t* A;
  const half_t* Bt;
  half_t* Oh = nullptr;
  float*  Of = nullptr;
  int ldc;
  if (MODE == 0) {
    int m = z >> 5, o = z & 31;
    A   = Abase + (size_t)m * CC * KD;
    Bt  = Bbase + (size_t)o * FF * KD;
    Oh  = (half_t*)Obase + (size_t)z * CC * FF;
    ldc = FF;
  } else if (MODE == 1) {
    int k = z >> 5, o = z & 31;
    A   = Abase + (size_t)((k >> 1) * OO + o) * CC * KD;
    Bt  = Bbase + (size_t)((k + 1) >> 1) * CC * KD;
    Of  = (float*)Obase + (size_t)z * CC * CC;
    ldc = CC;
  } else {
    A   = Abase;
    Bt  = Bbase;
    Of  = (float*)Obase;
    ldc = NPG;
  }
  const half_t* Ab = A + (size_t)mBase * KD;
  const half_t* Bb = Bt + (size_t)nBase * KD;

  // ---- async tile-copy setup: each thread owns rows (tid>>2) and (tid>>2)+64,
  //      16B segment (tid&3); offsets advance by 64B per k-step.
  const int rr0 = tid >> 2;
  const int rr1 = rr0 + 64;
  const int seg = tid & 3;
  int rbg0 = rr0, rbg1 = rr1;
  if (MODE == 2) {  // clamp source row; LDS destination row unchanged
    int n0 = nBase + rr0; if (n0 > NPG - 1) n0 = NPG - 1; rbg0 = n0 - nBase;
    int n1 = nBase + rr1; if (n1 > NPG - 1) n1 = NPG - 1; rbg1 = n1 - nBase;
  }
  const unsigned offA0 = (unsigned)((rr0 * KD + seg * 8) * 2);
  const unsigned offA1 = (unsigned)((rr1 * KD + seg * 8) * 2);
  const unsigned offB0 = (unsigned)((rbg0 * KD + seg * 8) * 2);
  const unsigned offB1 = (unsigned)((rbg1 * KD + seg * 8) * 2);
  const unsigned ldsA0 = (unsigned)(uintptr_t)&As[0][rr0][seg * 8];
  const unsigned ldsA1 = (unsigned)(uintptr_t)&As[0][rr1][seg * 8];
  const unsigned ldsB0 = (unsigned)(uintptr_t)&Bs[0][rr0][seg * 8];
  const unsigned ldsB1 = (unsigned)(uintptr_t)&Bs[0][rr1][seg * 8];
  const unsigned long long aS = (unsigned long long)(uintptr_t)Ab;
  const unsigned long long bS = (unsigned long long)(uintptr_t)Bb;

  auto issue = [&](unsigned bo, unsigned ko) {
    asm volatile("global_load_async_to_lds_b128 %0, %1, %2"
                 :: "v"(ldsA0 + bo), "v"(offA0 + ko), "s"(aS) : "memory");
    asm volatile("global_load_async_to_lds_b128 %0, %1, %2"
                 :: "v"(ldsA1 + bo), "v"(offA1 + ko), "s"(aS) : "memory");
    asm volatile("global_load_async_to_lds_b128 %0, %1, %2"
                 :: "v"(ldsB0 + bo), "v"(offB0 + ko), "s"(bS) : "memory");
    asm volatile("global_load_async_to_lds_b128 %0, %1, %2"
                 :: "v"(ldsB1 + bo), "v"(offB1 + ko), "s"(bS) : "memory");
  };

  v8f acc[4][2];
#pragma unroll
  for (int i = 0; i < 4; ++i)
#pragma unroll
    for (int j = 0; j < 2; ++j)
#pragma unroll
      for (int e = 0; e < 8; ++e) acc[i][j][e] = 0.0f;

  issue(0u, 0u);
  asm volatile("s_wait_asynccnt 0x0" ::: "memory");
  __syncthreads();

  union U { v16h v; uint4 q[2]; };
  int buf = 0;
  for (int kt = 0; kt < NK; ++kt) {
    if (kt + 1 < NK) issue((unsigned)(buf ^ 1) * BUFB, (unsigned)(kt + 1) * 64u);
    U bfr[2];
#pragma unroll
    for (int j = 0; j < 2; ++j) {
      int n = waveN * 32 + j * 16 + mrow;
      bfr[j].q[0] = *(const uint4*)&Bs[buf][n][sel * 8];
      bfr[j].q[1] = *(const uint4*)&Bs[buf][n][16 + sel * 8];
    }
#pragma unroll
    for (int i = 0; i < 4; ++i) {
      int r = waveM * 64 + i * 16 + mrow;
      U af;
      af.q[0] = *(const uint4*)&As[buf][r][sel * 8];
      af.q[1] = *(const uint4*)&As[buf][r][16 + sel * 8];
#pragma unroll
      for (int j = 0; j < 2; ++j) {
        acc[i][j] = __builtin_amdgcn_wmma_f32_16x16x32_f16(
            false, af.v, false, bfr[j].v, (short)0, acc[i][j], false, false);
      }
    }
    asm volatile("s_wait_asynccnt 0x0" ::: "memory");
    __syncthreads();
    buf ^= 1;
  }

  // D layout: VGPR v, lane -> (m = v + 8*sel, n = lane&15) within each 16x16 tile
#pragma unroll
  for (int i = 0; i < 4; ++i) {
    int rg = mBase + waveM * 64 + i * 16 + sel * 8;
#pragma unroll
    for (int j = 0; j < 2; ++j) {
      int cg = nBase + waveN * 32 + j * 16 + mrow;
      if (MODE == 2 && cg >= NPG) continue;
#pragma unroll
      for (int v = 0; v < 8; ++v) {
        size_t off = (size_t)(rg + v) * ldc + cg;
        if (MODE == 0) Oh[off] = (half_t)acc[i][j][v];
        else           Of[off] = acc[i][j][v];
      }
    }
  }
}

// ---------------------------------------------------------------------------
// Combine: y[b,k,r,c,o] = Ytmp[k,o,r,c] + P2[row_f, o*9+g] + P3[row_t, o*9+g]
//                         + T4[g*32+o]       (bias already in T4)
// Coalesced (c,o)-contiguous stores via LDS transpose. grid (128 r, 15 k).
// ---------------------------------------------------------------------------
__global__ __launch_bounds__(256)
void combine_kernel(const float* __restrict__ Ytmp, const float* __restrict__ P2b,
                    const float* __restrict__ P3b, const float* __restrict__ T4,
                    float* __restrict__ yout) {
  __shared__ float ybuf[32][130];
  __shared__ float p2row[NPG];
  __shared__ float t4s[NPG];
  const int t = threadIdx.x;
  const int k = blockIdx.y;
  const int r = blockIdx.x;
  const int mf = k >> 1, mt = (k + 1) >> 1;

  for (int j = t; j < NPG; j += 256) {
    p2row[j] = P2b[(size_t)(mf * CC + r) * NPG + j];
    t4s[j]   = T4[j];
  }
#pragma unroll
  for (int og = 0; og < 16; ++og) {
    int o = og * 2 + (t >> 7);
    int c = t & 127;
    ybuf[o][c] = Ytmp[((size_t)(k * OO + o) * CC + r) * CC + c];
  }
  __syncthreads();

  const int o = t & 31;
  float* ydst = yout + ((size_t)k * CC + r) * CC * OO;
#pragma unroll
  for (int cg = 0; cg < 16; ++cg) {
    int c = cg * 8 + (t >> 5);
    int x = (k & 1) ? (c + CC - r) : (c - r);
    int g = (x >= 2) + (x >= 4) + (x >= 8) + (x >= 16) +
            (x >= 32) + (x >= 64) + (x >= 128) + (x >= 256);
    float val = ybuf[o][c] + p2row[o * NG + g] +
                P3b[(size_t)(mt * CC + c) * NPG + o * NG + g] +
                t4s[g * OO + o];
    ydst[(size_t)c * OO + o] = val;
  }
}

// ---------------------------------------------------------------------------
// row_idx / col_idx (int32 bit-patterns into the float output buffer)
// ---------------------------------------------------------------------------
__global__ void idx_kernel(int* __restrict__ orow, int* __restrict__ ocol) {
  int i = blockIdx.x * 256 + threadIdx.x;
  if (i >= KK * CC * CC) return;
  int k = i / (CC * CC);
  int rc = i - k * (CC * CC);
  int r = rc >> 7, c = rc & 127;
  orow[i] = (k >> 1) * CC + r;
  ocol[i] = ((k + 1) >> 1) * CC + c;
}

// ---------------------------------------------------------------------------
extern "C" void kernel_launch(void* const* d_in, const int* in_sizes, int n_in,
                              void* d_out, int out_size, void* d_ws, size_t ws_size,
                              hipStream_t stream) {
  const float* from_x   = (const float*)d_in[0]; // (4,1024,768)
  const float* to_x     = (const float*)d_in[1]; // (4,1024,768)
  const float* weight   = (const float*)d_in[2]; // (32,788,788)
  const float* bias     = (const float*)d_in[3]; // (32,)
  const float* from_emb = (const float*)d_in[4]; // (9,20)
  const float* to_emb   = (const float*)d_in[5]; // (9,20)
  float* yout = (float*)d_out;                   // y then row_idx then col_idx

  char* p = (char*)d_ws;
  auto take = [&](size_t bytes) {
    char* r = p;
    p += (bytes + 255) & ~(size_t)255;
    return (void*)r;
  };
  const size_t nFX = (size_t)BB * LL * FF;
  half_t* Fh   = (half_t*)take(nFX * 2);
  half_t* Th   = (half_t*)take(nFX * 2);
  half_t* WffT = (half_t*)take((size_t)OO * FF * FF * 2);
  half_t* v2t  = (half_t*)take((size_t)NPG * FF * 2);
  half_t* v3t  = (half_t*)take((size_t)NPG * FF * 2);
  float*  T4   = (float*)take(NPG * 4);
  float*  P2   = (float*)take((size_t)BB * LL * NPG * 4);
  float*  P3   = (float*)take((size_t)BB * LL * NPG * 4);
  half_t* Gf   = (half_t*)take((size_t)NCH * OO * CC * FF * 2);   // per-b
  float*  Ytmp = (float*)take((size_t)KK * OO * CC * CC * 4);     // per-b

  // Stage 0: conversions and small precomputes
  convert_f16_kernel<<<2048, 256, 0, stream>>>(from_x, Fh, to_x, Th, nFX);
  wffT_kernel<<<dim3(24, 24, 32), 256, 0, stream>>>(weight, WffT);
  v2v3_kernel<<<(NPG * FF + 255) / 256, 256, 0, stream>>>(weight, from_emb, to_emb, v2t, v3t);
  t4_kernel<<<1, NPG, 0, stream>>>(weight, bias, from_emb, to_emb, T4);

  // P2 = from_x @ v2t^T, P3 = to_x @ v3t^T  (M=4096, N=288, K=768)
  gemm_tpl<2><<<dim3(3, 32, 1), 256, 0, stream>>>(Fh, v2t, P2);
  gemm_tpl<2><<<dim3(3, 32, 1), 256, 0, stream>>>(Th, v3t, P3);

  // Per-b main pipeline (caps workspace at ~142 MB)
  for (int b = 0; b < BB; ++b) {
    const half_t* Fhb = Fh + (size_t)b * LL * FF;
    const half_t* Thb = Th + (size_t)b * LL * FF;
    // Stage 1: Gf[m,o] = F_chunk[m] @ Wff[o]    (256 batched 128x768x768 GEMMs)
    gemm_tpl<0><<<dim3(6, 1, NCH * OO), 256, 0, stream>>>(Fhb, WffT, Gf);
    // Stage 2: Ytmp[k,o] = Gf[k/2,o] @ T_chunk[(k+1)/2]^T (480 batched 128x128x768)
    gemm_tpl<1><<<dim3(1, 1, KK * OO), 256, 0, stream>>>(Gf, Thb, Ytmp);
    // Stage 3: combine + transpose to (r,c,o)-contiguous output
    combine_kernel<<<dim3(CC, KK), 256, 0, stream>>>(
        Ytmp, P2 + (size_t)b * LL * NPG, P3 + (size_t)b * LL * NPG, T4,
        yout + (size_t)b * KK * CC * CC * OO);
  }

  // Index outputs (int32 values appended after y)
  const size_t ySz = (size_t)BB * KK * CC * CC * OO;
  const int nIdx = KK * CC * CC;
  idx_kernel<<<(nIdx + 255) / 256, 256, 0, stream>>>(
      (int*)(yout + ySz), (int*)(yout + ySz) + nIdx);
}
